// JaxTinyTextTower_82016695485038
// MI455X (gfx1250) — matrix operand order repaired and checked by reading, not compiled, not run on hardware
//
#include <hip/hip_runtime.h>

// ---------------------------------------------------------------------------
// Tiny Gemma-style text tower on MI455X (gfx1250, wave32, WMMA).
// - All GEMMs: v_wmma_f32_16x16x32_f16 (f16 operands, fp32 accum).
// - Weights pre-packed once per layer into f16 B-fragment layout (L2-resident,
//   each WMMA B-operand = 2x global_load_b128).
// - Flash attention stages K/V tiles with double-buffered
//   global_load_async_to_lds_b128 + s_wait_asynccnt (ASYNCcnt path).
// - Masks recomputed analytically; the 33MB mask inputs are never read.
// ---------------------------------------------------------------------------

#define BB 4
#define TT 2048
#define EMB 32
#define NH 4
#define NKV 2
#define HD 8
#define HID 64
#define NPLI 8
#define NL 4
#define EPS 1e-6f
#define MASKNEG -2.3819763e38f
#define NPACK 21            // weight fragment packs per layer (512 halves each)

typedef __attribute__((ext_vector_type(16))) _Float16 v16h;
typedef __attribute__((ext_vector_type(8)))  float    v8f;

__device__ __forceinline__ v8f wmma_f16(v16h a, v16h b, v8f c) {
  // D = A(16x32 f16) * B(32x16 f16) + C(16x16 f32)
  return __builtin_amdgcn_wmma_f32_16x16x32_f16(false, a, false, b, (short)0, c, false, false);
}

// 32-bit LDS offset of a __shared__ object (addrspace(3) ptrtoint).
typedef __attribute__((address_space(3))) void lds_void_t;
__device__ __forceinline__ unsigned lds_off_u32(void* p) {
  return (unsigned)(unsigned long long)(lds_void_t*)p;
}

// Per-lane async 16B copy global -> LDS (tracked by ASYNCcnt).
__device__ __forceinline__ void async_b128_to_lds(void* lds_dst, const void* gsrc) {
  unsigned loff = lds_off_u32(lds_dst);
  asm volatile("global_load_async_to_lds_b128 %0, %1, off"
               :: "v"(loff), "v"((unsigned long long)(size_t)gsrc)
               : "memory");
}

// A fragment (16-bit, 16x32 MxK) from LDS, row-major, leading dim ld.
__device__ __forceinline__ v16h load_A_lds(const _Float16* X, int ld, int lane) {
  int m = lane & 15, h2 = (lane >> 4) & 1;
  v16h a;
#pragma unroll
  for (int j = 0; j < 8; ++j) {
    int kk = h2 * 16 + 2 * j;
    a[2 * j]     = X[m * ld + kk];
    a[2 * j + 1] = X[m * ld + kk + 1];
  }
  return a;
}

// B fragment (16-bit, 32x16 KxN) from fp32 global weights W[k*ldn + n0+n].
// Zero-pads columns >= nmax and rows >= kmax.  (Used only in the prep kernel.)
__device__ __forceinline__ v16h load_B_f32(const float* W, int ldn, int n0,
                                           int nmax, int kmax, int lane) {
  int n = lane & 15, kb = ((lane >> 4) & 1) * 16;
  bool nok = (n0 + n) < nmax;
  v16h b;
#pragma unroll
  for (int j = 0; j < 8; ++j) {
    int k0 = kb + 2 * j;
    b[2 * j]     = (nok && (k0     < kmax)) ? (_Float16)W[(k0    ) * ldn + n0 + n] : (_Float16)0.f;
    b[2 * j + 1] = (nok && (k0 + 1 < kmax)) ? (_Float16)W[(k0 + 1) * ldn + n0 + n] : (_Float16)0.f;
  }
  return b;
}

// Pre-packed B fragment: 32 lanes x 16 halves, per-lane contiguous (32B).
__device__ __forceinline__ v16h load_pack(const _Float16* wf, int lane) {
  return *(const v16h*)(wf + lane * 16);
}

// C/D fragment (fp32 16x16) -> LDS row-major with leading dim ld.
__device__ __forceinline__ void store_D_lds(float* Y, int ld, v8f d, int lane) {
  int n = lane & 15, mb = ((lane >> 4) & 1) * 8;
#pragma unroll
  for (int j = 0; j < 8; ++j) Y[(mb + j) * ld + n] = d[j];
}

__device__ __forceinline__ float gelu_tanh(float x) {
  float x3 = x * x * x;
  return 0.5f * x * (1.f + tanhf(0.7978845608028654f * (x + 0.044715f * x3)));
}

// dst[m][:] += rmsnorm(src[m][:], scale) over EMB=32. Two lanes per token.
__device__ __forceinline__ void add_rmsnorm32(float* dst, const float* src,
                                              const float* scale, int lane) {
  int m = lane & 15, e0 = ((lane >> 4) & 1) * 16;
  float s = 0.f;
#pragma unroll
  for (int e = 0; e < 32; ++e) { float x = src[m * 32 + e]; s += x * x; }
  float r = rsqrtf(s * (1.f / 32.f) + EPS);
  for (int e = e0; e < e0 + 16; ++e)
    dst[m * 32 + e] += src[m * 32 + e] * r * (1.f + scale[e]);
}

// dst[m][:] = (f16) rmsnorm(src[m][:], scale) over EMB=32.
__device__ __forceinline__ void rmsnorm32_to_f16(_Float16* dst, int ld, const float* src,
                                                 const float* scale, int lane) {
  int m = lane & 15, e0 = ((lane >> 4) & 1) * 16;
  float s = 0.f;
#pragma unroll
  for (int e = 0; e < 32; ++e) { float x = src[m * 32 + e]; s += x * x; }
  float r = rsqrtf(s * (1.f / 32.f) + EPS);
  for (int e = e0; e < e0 + 16; ++e)
    dst[m * ld + e] = (_Float16)(src[m * 32 + e] * r * (1.f + scale[e]));
}

// ---------------------------------------------------------------------------
// K0: pre-pack one layer's weights into f16 B-fragment packs (one wave/pack).
// Pack map: 0-1 wq | 2 wk | 3 wv | 4-5 wo | 6-9 wg | 10-13 wu |
//           14-17 wd (kchunk*2+ntile) | 18 pli_gate | 19-20 pli_proj
// ---------------------------------------------------------------------------
__global__ void k_prep_weights(const float* __restrict__ wq, const float* __restrict__ wk,
                               const float* __restrict__ wv, const float* __restrict__ wo,
                               const float* __restrict__ wg, const float* __restrict__ wu,
                               const float* __restrict__ wd, const float* __restrict__ pli_gate,
                               const float* __restrict__ pli_proj, _Float16* __restrict__ wf) {
  int p = blockIdx.x, lane = threadIdx.x;
  const float* W; int ldn, n0, nmax, kmax = 32;
  if (p < 2)        { W = wq;       ldn = 32; n0 = p * 16;           nmax = 32; }
  else if (p == 2)  { W = wk;       ldn = 16; n0 = 0;                nmax = 16; }
  else if (p == 3)  { W = wv;       ldn = 16; n0 = 0;                nmax = 16; }
  else if (p < 6)   { W = wo;       ldn = 32; n0 = (p - 4) * 16;     nmax = 32; }
  else if (p < 10)  { W = wg;       ldn = 64; n0 = (p - 6) * 16;     nmax = 64; }
  else if (p < 14)  { W = wu;       ldn = 64; n0 = (p - 10) * 16;    nmax = 64; }
  else if (p < 18)  { W = wd + ((p - 14) >> 1) * 32 * 32;
                                    ldn = 32; n0 = ((p - 14) & 1) * 16; nmax = 32; }
  else if (p == 18) { W = pli_gate; ldn = 8;  n0 = 0;                nmax = 8;  }
  else              { W = pli_proj; ldn = 32; n0 = (p - 19) * 16;    nmax = 32; kmax = 8; }
  v16h b = load_B_f32(W, ldn, n0, nmax, kmax, lane);
  *(v16h*)(wf + (size_t)p * 512 + lane * 16) = b;
}

// ---------------------------------------------------------------------------
// K1: embedding + per-layer-input projection.  One token per 32-thread block.
// ---------------------------------------------------------------------------
__global__ void k_embed(const int* __restrict__ tokens, const float* __restrict__ embed,
                        const float* __restrict__ pl_embed, const float* __restrict__ pl_model_proj,
                        float* __restrict__ h, float* __restrict__ pli) {
  int idx = blockIdx.x;          // token index in [0, B*T)
  int lane = threadIdx.x;        // 0..31
  int tok = tokens[idx];
  __shared__ float hs[32];
  float hv = embed[tok * 32 + lane] * 5.656854249492381f;   // sqrt(32)
  hs[lane] = hv;
  h[idx * 32 + lane] = hv;
  __syncthreads();
  float acc = 0.f;
#pragma unroll
  for (int e = 0; e < 32; ++e) acc += hs[e] * pl_model_proj[e * 32 + lane];
  float pe = pl_embed[tok * 32 + lane] * 2.8284271247461903f;  // sqrt(8)
  pli[idx * 32 + lane] = (acc * 0.17677669529663687f + pe) * 0.7071067811865476f;
}

// ---------------------------------------------------------------------------
// K2: fused pre-attn RMSNorm + QKV GEMMs (WMMA on packed weights) + qk-norm
// + RoPE.  One wave per 16-token tile; ROT is compile-time rope width.
// ---------------------------------------------------------------------------
template <int ROT>
__global__ void k_qkv(const float* __restrict__ h, const int* __restrict__ positions,
                      const _Float16* __restrict__ wf,
                      const float* __restrict__ q_scale, const float* __restrict__ k_scale,
                      const float* __restrict__ pre_attn, float rope_base,
                      _Float16* __restrict__ qo, _Float16* __restrict__ ko,
                      _Float16* __restrict__ vo) {
  int lane = threadIdx.x;
  int tile = blockIdx.x;
  int t0 = (tile & 127) * 16;
  int b = tile >> 7;
  int base = (b * TT + t0) * 32;

  __shared__ float    hs[16 * 32];
  __shared__ _Float16 xh[16 * 32];
  __shared__ float    qs[16 * 32];
  __shared__ float    ks[16 * 16];
  __shared__ float    vs[16 * 16];

  for (int i = lane; i < 512; i += 32) hs[i] = h[base + i];
  __syncthreads();
  rmsnorm32_to_f16(xh, 32, hs, pre_attn, lane);
  __syncthreads();

  v8f z = {};
  v16h a = load_A_lds(xh, 32, lane);
  v8f dq0 = wmma_f16(a, load_pack(wf + 0 * 512, lane), z);
  v8f dq1 = wmma_f16(a, load_pack(wf + 1 * 512, lane), z);
  v8f dk  = wmma_f16(a, load_pack(wf + 2 * 512, lane), z);
  v8f dv  = wmma_f16(a, load_pack(wf + 3 * 512, lane), z);
  store_D_lds(qs,      32, dq0, lane);
  store_D_lds(qs + 16, 32, dq1, lane);
  store_D_lds(ks, 16, dk, lane);
  store_D_lds(vs, 16, dv, lane);
  __syncthreads();

  constexpr int HALF = ROT / 2;
  // Q: 64 (token,head) groups, 2 per lane.
  for (int g = lane; g < 64; g += 32) {
    int m = g >> 2, hd = g & 3;
    const float* qq = &qs[m * 32 + hd * 8];
    float s = 0.f;
#pragma unroll
    for (int d = 0; d < 8; ++d) s += qq[d] * qq[d];
    float r = rsqrtf(s * 0.125f + EPS);
    float vreg[8];
#pragma unroll
    for (int d = 0; d < 8; ++d) vreg[d] = qq[d] * r * (1.f + q_scale[d]);
    float pos = (float)positions[b * TT + t0 + m];
#pragma unroll
    for (int i = 0; i < HALF; ++i) {
      float ts = __powf(rope_base, 2.f * (float)i / (float)ROT);
      float ang = pos / ts;
      float sn = __sinf(ang), cs = __cosf(ang);
      float x0 = vreg[i], x1 = vreg[i + HALF];
      vreg[i]        = x0 * cs - x1 * sn;
      vreg[i + HALF] = x1 * cs + x0 * sn;
    }
#pragma unroll
    for (int d = 0; d < 8; ++d)
      qo[((size_t)(b * TT + t0 + m) * NH + hd) * HD + d] =
          (_Float16)(vreg[d] * 0.3535533905932738f);   // HEAD_DIM^-0.5
  }
  // K: 32 (token,kv) groups, 1 per lane.
  {
    int mm = lane >> 1, kv = lane & 1;
    const float* kk = &ks[mm * 16 + kv * 8];
    float s = 0.f;
#pragma unroll
    for (int d = 0; d < 8; ++d) s += kk[d] * kk[d];
    float r = rsqrtf(s * 0.125f + EPS);
    float vreg[8];
#pragma unroll
    for (int d = 0; d < 8; ++d) vreg[d] = kk[d] * r * (1.f + k_scale[d]);
    float pos = (float)positions[b * TT + t0 + mm];
#pragma unroll
    for (int i = 0; i < HALF; ++i) {
      float ts = __powf(rope_base, 2.f * (float)i / (float)ROT);
      float ang = pos / ts;
      float sn = __sinf(ang), cs = __cosf(ang);
      float x0 = vreg[i], x1 = vreg[i + HALF];
      vreg[i]        = x0 * cs - x1 * sn;
      vreg[i + HALF] = x1 * cs + x0 * sn;
    }
#pragma unroll
    for (int d = 0; d < 8; ++d)
      ko[((size_t)(b * TT + t0 + mm) * NKV + kv) * HD + d] = (_Float16)vreg[d];
  }
  // V passthrough to f16.
  for (int i = lane; i < 256; i += 32) {
    int m = i >> 4, c = i & 15;
    vo[(size_t)(b * TT + t0 + m) * 16 + c] = (_Float16)vs[i];
  }
}

// ---------------------------------------------------------------------------
// K3: flash attention, one wave per (b, head, 16-query tile).
// 32 keys/step: K/V tiles staged by double-buffered async LDS copies,
// 2 score WMMAs + 1 P*V WMMA, fp32 online softmax.
// WIN==0 -> global causal; WIN>0 -> sliding window.
// ---------------------------------------------------------------------------
template <int WIN>
__global__ void k_attn(const _Float16* __restrict__ qo, const _Float16* __restrict__ ko,
                       const _Float16* __restrict__ vo, float* __restrict__ attn) {
  int lane = threadIdx.x;
  int id = blockIdx.x;
  int qt = id & 127;
  int head = (id >> 7) & 3;
  int b = id >> 9;
  int kvh = head >> 1;                      // GQA: heads {0,1}->kv0, {2,3}->kv1
  int t0 = qt * 16;

  __shared__ _Float16 pa[16 * 32];                      // probs for A fragment
  __shared__ __align__(16) _Float16 kbuf[2][32 * 8];    // staged K   [key][dim]
  __shared__ __align__(16) _Float16 vbuf[2][32 * 8];    // staged V   [key][dim]

  int m = lane & 15;
  int h2 = (lane >> 4) & 1;
  int n = lane & 15;

  // Q as A fragment: K-dim = head_dim (8 valid, rest zero-padded).
  v16h aq;
  {
    const _Float16* qp = qo + ((size_t)(b * TT + t0 + m) * NH + head) * HD;
#pragma unroll
    for (int j = 0; j < 8; ++j) {
      int kk = h2 * 16 + 2 * j;
      aq[2 * j]     = (kk     < 8) ? qp[kk]     : (_Float16)0.f;
      aq[2 * j + 1] = (kk + 1 < 8) ? qp[kk + 1] : (_Float16)0.f;
    }
  }

  float M[8], L[8];
  v8f acc = {};
#pragma unroll
  for (int j = 0; j < 8; ++j) { M[j] = -3.0e38f; L[j] = 0.f; }

  int ktstart = 0;
  if (WIN > 0) { int lo = t0 - (WIN - 1); if (lo < 0) lo = 0; ktstart = lo >> 4; }

  // Stage the first 32-key block (clamped token -> always in-bounds; masked
  // keys get p==0 so clamped garbage never contributes).
  {
    int tok = ktstart * 16 + lane; if (tok > TT - 1) tok = TT - 1;
    async_b128_to_lds(&kbuf[0][lane * 8], ko + ((size_t)(b * TT + tok) * NKV + kvh) * HD);
    async_b128_to_lds(&vbuf[0][lane * 8], vo + (size_t)(b * TT + tok) * 16 + kvh * 8);
  }

  int nb = 0;
  for (int kt = ktstart; kt <= qt; kt += 2) {
    int kb0 = kt * 16;
    bool hasnext = (kt + 2 <= qt);      // uniform across the wave
    if (hasnext) {                       // prefetch next block into other buffer
      int tok = kb0 + 32 + lane; if (tok > TT - 1) tok = TT - 1;
      async_b128_to_lds(&kbuf[nb ^ 1][lane * 8], ko + ((size_t)(b * TT + tok) * NKV + kvh) * HD);
      async_b128_to_lds(&vbuf[nb ^ 1][lane * 8], vo + (size_t)(b * TT + tok) * 16 + kvh * 8);
      asm volatile("s_wait_asynccnt 0x2" ::: "memory");   // current block done
    } else {
      asm volatile("s_wait_asynccnt 0x0" ::: "memory");
    }
    const _Float16* kb_ = kbuf[nb];
    const _Float16* vb_ = vbuf[nb];

    // K^T as B fragments: rows = dim (8 valid), cols = keys (from LDS).
    v16h bk0, bk1;
    {
      int kb = h2 * 16;
#pragma unroll
      for (int j = 0; j < 8; ++j) {
        int r0 = kb + 2 * j, r1 = r0 + 1;
        bk0[2 * j]     = (r0 < 8) ? kb_[n * 8 + r0]        : (_Float16)0.f;
        bk0[2 * j + 1] = (r1 < 8) ? kb_[n * 8 + r1]        : (_Float16)0.f;
        bk1[2 * j]     = (r0 < 8) ? kb_[(16 + n) * 8 + r0] : (_Float16)0.f;
        bk1[2 * j + 1] = (r1 < 8) ? kb_[(16 + n) * 8 + r1] : (_Float16)0.f;
      }
    }
    v8f z = {};
    v8f s0 = wmma_f16(aq, bk0, z);
    v8f s1 = wmma_f16(aq, bk1, z);

    // mask + online softmax (row stats via intra-half shuffles).
#pragma unroll
    for (int j = 0; j < 8; ++j) {
      int qrow = t0 + j + 8 * h2;
      int key0 = kb0 + n, key1 = kb0 + 16 + n;
      bool v0 = (key0 <= qrow) && (WIN == 0 || (qrow - key0) < WIN);
      bool v1 = (key1 <= qrow) && (WIN == 0 || (qrow - key1) < WIN) && (key1 < TT);
      float x0 = v0 ? s0[j] : MASKNEG;
      float x1 = v1 ? s1[j] : MASKNEG;
      float mx = fmaxf(x0, x1);
#pragma unroll
      for (int off = 1; off < 16; off <<= 1) mx = fmaxf(mx, __shfl_xor(mx, off, 32));
      float newM = fmaxf(M[j], mx);
      float alpha = __expf(M[j] - newM);
      float e0 = __expf(x0 - newM);
      float e1 = __expf(x1 - newM);
      float rs = e0 + e1;
#pragma unroll
      for (int off = 1; off < 16; off <<= 1) rs += __shfl_xor(rs, off, 32);
      L[j] = L[j] * alpha + rs;
      M[j] = newM;
      acc[j] *= alpha;
      pa[(j + 8 * h2) * 32 + n]      = (_Float16)e0;
      pa[(j + 8 * h2) * 32 + 16 + n] = (_Float16)e1;
    }
    __syncthreads();
    v16h ap = load_A_lds(pa, 32, lane);
    // V as B fragment: rows = key-local (32), cols = dim (8 valid, from LDS).
    v16h bv;
    {
      int kb = h2 * 16;
      bool nok = n < 8;
#pragma unroll
      for (int j = 0; j < 8; ++j) {
        int kr0 = kb + 2 * j;
        bv[2 * j]     = nok ? vb_[kr0 * 8 + n]       : (_Float16)0.f;
        bv[2 * j + 1] = nok ? vb_[(kr0 + 1) * 8 + n] : (_Float16)0.f;
      }
    }
    acc = wmma_f16(ap, bv, acc);
    __syncthreads();
    nb ^= 1;
  }

  if (n < 8) {
#pragma unroll
    for (int j = 0; j < 8; ++j) {
      int t = t0 + j + 8 * h2;
      attn[(size_t)(b * TT + t) * 32 + head * 8 + n] = acc[j] / L[j];
    }
  }
}

// ---------------------------------------------------------------------------
// K4: fused block tail: wo proj + post-attn norm residual, FFW (gelu-gated)
// + post-ffw norm residual, per-layer-input gate/proj + post-pli residual.
// One wave per 16-token tile; all GEMMs via WMMA on packed weights.
// ---------------------------------------------------------------------------
__global__ void k_block_tail(float* __restrict__ h, const float* __restrict__ pli,
                             const float* __restrict__ attn,
                             const _Float16* __restrict__ wf,
                             const float* __restrict__ post_attn,
                             const float* __restrict__ pre_ffw,
                             const float* __restrict__ post_ffw,
                             const float* __restrict__ post_pli, int layer_idx) {
  int lane = threadIdx.x;
  int tile = blockIdx.x;
  int t0 = (tile & 127) * 16;
  int b = tile >> 7;
  int base = (b * TT + t0) * 32;

  __shared__ float    hs[16 * 32];
  __shared__ float    st[16 * 32];
  __shared__ float    gbuf[16 * 64];
  __shared__ float    ubuf[16 * 64];
  __shared__ _Float16 ah[16 * 64];

  for (int i = lane; i < 512; i += 32) hs[i] = h[base + i];
  for (int i = lane; i < 512; i += 32) ah[i] = (_Float16)attn[base + i];
  __syncthreads();

  v8f z = {};
  {   // a = attn @ wo   (packs 4,5)
    v16h a = load_A_lds(ah, 32, lane);
    v8f d0 = wmma_f16(a, load_pack(wf + 4 * 512, lane), z);
    v8f d1 = wmma_f16(a, load_pack(wf + 5 * 512, lane), z);
    store_D_lds(st,      32, d0, lane);
    store_D_lds(st + 16, 32, d1, lane);
  }
  __syncthreads();
  add_rmsnorm32(hs, st, post_attn, lane);       // h += rmsnorm(a, post_attn)
  __syncthreads();
  rmsnorm32_to_f16(ah, 32, hs, pre_ffw, lane);  // f = rmsnorm(h, pre_ffw)
  __syncthreads();
  {   // g = f@wg (packs 6..9), u = f@wu (packs 10..13)
    v16h a = load_A_lds(ah, 32, lane);
#pragma unroll
    for (int nt = 0; nt < 4; ++nt) {
      v8f dg = wmma_f16(a, load_pack(wf + (6 + nt) * 512, lane), z);
      v8f du = wmma_f16(a, load_pack(wf + (10 + nt) * 512, lane), z);
      store_D_lds(gbuf + nt * 16, 64, dg, lane);
      store_D_lds(ubuf + nt * 16, 64, du, lane);
    }
  }
  __syncthreads();
  for (int i = lane; i < 1024; i += 32)
    ah[i] = (_Float16)(gelu_tanh(gbuf[i]) * ubuf[i]);
  __syncthreads();
  {   // ff = act @ wd  (packs 14..17: kchunk*2 + ntile)
    v16h a0 = load_A_lds(ah,      64, lane);
    v16h a1 = load_A_lds(ah + 32, 64, lane);
#pragma unroll
    for (int nt = 0; nt < 2; ++nt) {
      v8f d = wmma_f16(a0, load_pack(wf + (14 + nt) * 512, lane), z);
      d     = wmma_f16(a1, load_pack(wf + (16 + nt) * 512, lane), d);
      store_D_lds(st + nt * 16, 32, d, lane);
    }
  }
  __syncthreads();
  add_rmsnorm32(hs, st, post_ffw, lane);        // h += rmsnorm(ff, post_ffw)
  __syncthreads();
  for (int i = lane; i < 512; i += 32) ah[i] = (_Float16)hs[i];
  __syncthreads();
  {   // gate = h @ pli_gate  (pack 18, N=8 zero-padded)
    v16h a = load_A_lds(ah, 32, lane);
    v8f d = wmma_f16(a, load_pack(wf + 18 * 512, lane), z);
    store_D_lds(st, 32, d, lane);
  }
  __syncthreads();
  for (int i = lane; i < 512; i += 32) {        // gact = gelu(gate)*pli, K-pad
    int mm = i >> 5, p = i & 31;
    float gv = 0.f;
    if (p < 8)
      gv = gelu_tanh(st[mm * 32 + p]) * pli[((size_t)(b * TT + t0 + mm) * NL + layer_idx) * NPLI + p];
    ah[i] = (_Float16)gv;
  }
  __syncthreads();
  {   // g = gact @ pli_proj  (packs 19,20; K=8 padded to 32)
    v16h a = load_A_lds(ah, 32, lane);
    v8f d0 = wmma_f16(a, load_pack(wf + 19 * 512, lane), z);
    v8f d1 = wmma_f16(a, load_pack(wf + 20 * 512, lane), z);
    store_D_lds(st,      32, d0, lane);
    store_D_lds(st + 16, 32, d1, lane);
  }
  __syncthreads();
  add_rmsnorm32(hs, st, post_pli, lane);        // h += rmsnorm(g, post_pli)
  __syncthreads();
  for (int i = lane; i < 512; i += 32) h[base + i] = hs[i];
}

// ---------------------------------------------------------------------------
// K5: final RMS norm -> output (fp32).
// ---------------------------------------------------------------------------
__global__ void k_final(const float* __restrict__ h, const float* __restrict__ final_norm,
                        float* __restrict__ out) {
  int idx = blockIdx.x;
  int lane = threadIdx.x;
  float x = h[idx * 32 + lane];
  float s = x * x;
#pragma unroll
  for (int off = 1; off < 32; off <<= 1) s += __shfl_xor(s, off, 32);
  float r = rsqrtf(s * (1.f / 32.f) + EPS);
  out[idx * 32 + lane] = x * r * (1.f + final_norm[lane]);
}

// ---------------------------------------------------------------------------
// Host-side launch. Input order = setup_inputs() insertion order:
//   0 tokens, 1 positions, 2 full_mask (unused), 3 sliding_mask (unused),
//   4 embed, 5 pl_embed, 6 pl_model_proj, 7 final_norm,
//   8.. per-layer (16 each): wq wk wv wo q_scale k_scale pre_attn post_attn
//                            pre_ffw post_ffw wg wu wd pli_gate pli_proj post_pli
// ---------------------------------------------------------------------------
extern "C" void kernel_launch(void* const* d_in, const int* in_sizes, int n_in,
                              void* d_out, int out_size, void* d_ws, size_t ws_size,
                              hipStream_t stream) {
  (void)in_sizes; (void)n_in; (void)out_size; (void)ws_size;
  const int* tokens    = (const int*)d_in[0];
  const int* positions = (const int*)d_in[1];
  const float* embed         = (const float*)d_in[4];
  const float* pl_embed      = (const float*)d_in[5];
  const float* pl_model_proj = (const float*)d_in[6];
  const float* final_norm    = (const float*)d_in[7];

  // workspace: h / pli / attn (fp32) + q / k / v (f16) + packed weights
  float* h    = (float*)d_ws;
  float* pli  = h   + (size_t)BB * TT * 32;
  float* attn = pli + (size_t)BB * TT * 32;
  _Float16* q  = (_Float16*)(attn + (size_t)BB * TT * 32);
  _Float16* k  = q + (size_t)BB * TT * 32;
  _Float16* v  = k + (size_t)BB * TT * 16;
  _Float16* wf = v + (size_t)BB * TT * 16;      // NL * NPACK * 512 halves

  // Pre-pack all layers' weights into f16 B-fragment layout.
  for (int l = 0; l < NL; ++l) {
    void* const* p = d_in + 8 + l * 16;
    k_prep_weights<<<NPACK, 32, 0, stream>>>(
        (const float*)p[0], (const float*)p[1], (const float*)p[2], (const float*)p[3],
        (const float*)p[10], (const float*)p[11], (const float*)p[12],
        (const float*)p[13], (const float*)p[14], wf + (size_t)l * NPACK * 512);
  }

  k_embed<<<BB * TT, 32, 0, stream>>>(tokens, embed, pl_embed, pl_model_proj, h, pli);

  const float bases[NL] = {1e4f, 1e4f, 1e4f, 1e6f};
  for (int l = 0; l < NL; ++l) {
    void* const* p = d_in + 8 + l * 16;
    const float* q_scale  = (const float*)p[4];
    const float* k_scale  = (const float*)p[5];
    const float* pre_attn = (const float*)p[6];
    const float* post_attn= (const float*)p[7];
    const float* pre_ffw  = (const float*)p[8];
    const float* post_ffw = (const float*)p[9];
    const float* post_pli = (const float*)p[15];
    const _Float16* lwf = wf + (size_t)l * NPACK * 512;

    dim3 gtile(BB * (TT / 16));
    if (l < 3) {
      k_qkv<8><<<gtile, 32, 0, stream>>>(h, positions, lwf, q_scale, k_scale,
                                         pre_attn, bases[l], q, k, v);
      k_attn<8><<<BB * NH * (TT / 16), 32, 0, stream>>>(q, k, v, attn);
    } else {
      k_qkv<2><<<gtile, 32, 0, stream>>>(h, positions, lwf, q_scale, k_scale,
                                         pre_attn, bases[l], q, k, v);
      k_attn<0><<<BB * NH * (TT / 16), 32, 0, stream>>>(q, k, v, attn);
    }
    k_block_tail<<<gtile, 32, 0, stream>>>(h, pli, attn, lwf, post_attn, pre_ffw,
                                           post_ffw, post_pli, l);
  }
  k_final<<<BB * TT, 32, 0, stream>>>(h, final_norm, (float*)d_out);
}
